// Gumbel_Softmax_2568390443494
// MI455X (gfx1250) — compile-verified, matrix operand order checked by reading, not run on hardware
//
#include <hip/hip_runtime.h>
#include <hip/hip_bf16.h>

typedef __attribute__((ext_vector_type(16))) _Float16 v16h;
typedef __attribute__((ext_vector_type(8)))  _Float16 h8;
typedef __attribute__((ext_vector_type(8)))  float    v8f;

#define BQ   8
#define CCH  256
#define NSP  4096        // H*W
#define KCW  512
#define NROWS (BQ * NSP) // 32768

#define SEL_SIZE  (BQ * CCH * NSP)        // 8388608
#define CW_OFF    SEL_SIZE                // 8388608
#define KLD_OFF   (CW_OFF + NROWS)        // 8421376
#define MIP_OFF   (KLD_OFF + 1)           // 8421377

// workspace byte offsets
#define WS_SF_OFF     0                       // 32768 f32
#define WS_SW_OFF     131072                  // 512 f32
#define WS_CBT_OFF    133120                  // 256*512 f16 (C-major, K contiguous)
#define WS_SOFT_OFF   395264                  // 32768*512 f16 row-major

__device__ __forceinline__ float wred_max(float v) {
#pragma unroll
    for (int m = 16; m > 0; m >>= 1) v = fmaxf(v, __shfl_xor(v, m, 32));
    return v;
}
__device__ __forceinline__ float wred_sum(float v) {
#pragma unroll
    for (int m = 16; m > 0; m >>= 1) v += __shfl_xor(v, m, 32);
    return v;
}

// ---- Kernel 0a: sf[b,n] = sum_c feats[b,c,n] ------------------------------
__global__ void sf_kernel(const float* __restrict__ feats, float* __restrict__ sf) {
    int idx = blockIdx.x * blockDim.x + threadIdx.x;   // 0..32767
    int b = idx >> 12, n = idx & (NSP - 1);
    const float* p = feats + (size_t)b * CCH * NSP + n;
    float s = 0.f;
#pragma unroll 8
    for (int c = 0; c < CCH; ++c) s += p[(size_t)c * NSP];
    sf[idx] = s;
}

// ---- Kernel 0b: sw[k] = sum_c codebook[k,c]; cbT[c,k] = (f16)codebook[k,c]
__global__ void cb_kernel(const float* __restrict__ codebook,
                          float* __restrict__ sw, _Float16* __restrict__ cbT,
                          float* __restrict__ kld_slot) {
    int k = blockIdx.x;          // 0..511
    int c = threadIdx.x;         // 0..255
    float v = codebook[(size_t)k * CCH + c];
    cbT[(size_t)c * KCW + k] = (_Float16)v;
    __shared__ float red[256];
    red[c] = v;
    __syncthreads();
#pragma unroll
    for (int s = 128; s > 0; s >>= 1) {
        if (c < s) red[c] += red[c + s];
        __syncthreads();
    }
    if (c == 0) sw[k] = red[0];
    if (k == 0 && c == 0) *kld_slot = 0.0f;
}

// ---- Kernel 1: per-token-row softmax / argmax / KLD / fp16 staging --------
// one wave32 per row; each lane owns 16 consecutive k's
__global__ void rows_kernel(const float* __restrict__ gumbel,
                            const float* __restrict__ sf,
                            const float* __restrict__ sw,
                            _Float16* __restrict__ soft16,
                            float* __restrict__ out) {
    __shared__ float sws[KCW];
    for (int i = threadIdx.x; i < KCW; i += blockDim.x) sws[i] = sw[i];
    __syncthreads();

    int wave = threadIdx.x >> 5;
    int lane = threadIdx.x & 31;
    int row  = blockIdx.x * 8 + wave;                 // 0..32767
    int b = row >> 12, n = row & (NSP - 1);
    float sfv = sf[row];

    const float4* g4 = (const float4*)(gumbel + (size_t)row * KCW + lane * 16);
    float g[16];
#pragma unroll
    for (int q = 0; q < 4; ++q) {
        float4 t = g4[q];
        g[q * 4 + 0] = t.x; g[q * 4 + 1] = t.y; g[q * 4 + 2] = t.z; g[q * 4 + 3] = t.w;
    }

    float logit[16], t[16];
    float tmax = -3.4e38f;
#pragma unroll
    for (int j = 0; j < 16; ++j) {
        float d = sfv - sws[lane * 16 + j];
        float l = __expf(-d * d);
        logit[j] = l;
        float tv = (l + g[j]) * 100.0f;               // /tau, tau=0.01
        t[j] = tv;
        tmax = fmaxf(tmax, tv);
    }
    tmax = wred_max(tmax);
    float ssum = 0.f;
#pragma unroll
    for (int j = 0; j < 16; ++j) { t[j] = __expf(t[j] - tmax); ssum += t[j]; }
    ssum = wred_sum(ssum);
    float inv = 1.0f / ssum;

    // stage fp16 probs (contiguous, WMMA A operand source)
    h8 lo, hi;
#pragma unroll
    for (int j = 0; j < 8; ++j) { lo[j] = (_Float16)(t[j] * inv); hi[j] = (_Float16)(t[j + 8] * inv); }
    _Float16* sp = soft16 + (size_t)row * KCW + lane * 16;
    *(h8*)(sp)     = lo;
    *(h8*)(sp + 8) = hi;

    // mat_id_probs (B,K,H,W): transposed scatter
    float* mip = out + MIP_OFF + (size_t)b * KCW * NSP + n;
#pragma unroll
    for (int j = 0; j < 16; ++j) mip[(size_t)(lane * 16 + j) * NSP] = t[j] * inv;

    // argmax with first-max tie-break
    float mv = -3.4e38f; int mi = 0;
#pragma unroll
    for (int j = 0; j < 16; ++j) {
        if (t[j] > mv) { mv = t[j]; mi = lane * 16 + j; }
    }
#pragma unroll
    for (int m = 16; m > 0; m >>= 1) {
        float ov = __shfl_xor(mv, m, 32);
        int   oi = __shfl_xor(mi, m, 32);
        if (ov > mv || (ov == mv && oi < mi)) { mv = ov; mi = oi; }
    }
    if (lane == 0) out[CW_OFF + row] = (float)mi;

    // qy = softmax(logits); kld contribution (logits <= 1 so exp is safe)
    float qs = 0.f, e[16];
#pragma unroll
    for (int j = 0; j < 16; ++j) { e[j] = __expf(logit[j]); qs += e[j]; }
    qs = wred_sum(qs);
    float qinv = 1.0f / qs, kl = 0.f;
#pragma unroll
    for (int j = 0; j < 16; ++j) {
        float qy = e[j] * qinv;
        kl += qy * __logf(qy * (float)KCW + 1e-10f);
    }
    kl = wred_sum(kl);
    if (lane == 0) atomicAdd(out + KLD_OFF, kl * (1.0f / (float)NROWS));
}

// ---- Kernel 2: WMMA GEMM: sel[b,c,n] = sum_k soft[row,k] * cb[k,c] --------
// block = 512 threads = 16 waves; block per 16-row M tile; wave per 16-col N tile
__global__ void gemm_kernel(const _Float16* __restrict__ soft16,
                            const _Float16* __restrict__ cbT,
                            float* __restrict__ out) {
    int wave = threadIdx.x >> 5;          // 0..15  -> n-tile
    int lane = threadIdx.x & 31;
    int m0 = blockIdx.x * 16;
    int n0 = wave * 16;
    int hi  = lane >> 4;                  // lane half
    int mr  = lane & 15;                  // row (A) / col (B) within tile

    // 16-bit A layout: lanes 0-15 hold K {8h+0..7, 8h+16..23}; M = lane%16
    const _Float16* aBase = soft16 + (size_t)(m0 + mr) * KCW + hi * 8;
    const _Float16* bBase = cbT    + (size_t)(n0 + mr) * KCW + hi * 8;

    v8f acc = {};
#pragma unroll
    for (int kk = 0; kk < 16; ++kk) {
        const _Float16* pa = aBase + kk * 32;
        const _Float16* pb = bBase + kk * 32;
        h8 a0 = *(const h8*)(pa);
        h8 a1 = *(const h8*)(pa + 16);
        h8 b0 = *(const h8*)(pb);
        h8 b1 = *(const h8*)(pb + 16);
        v16h A = __builtin_shufflevector(a0, a1, 0,1,2,3,4,5,6,7,8,9,10,11,12,13,14,15);
        v16h Bf = __builtin_shufflevector(b0, b1, 0,1,2,3,4,5,6,7,8,9,10,11,12,13,14,15);
        acc = __builtin_amdgcn_wmma_f32_16x16x32_f16(
            false, A, false, Bf, (short)0, acc, false, false);
    }

    // D layout: VGPR r -> M = r + 8*hi ; N = lane%16
    int colC = n0 + mr;
#pragma unroll
    for (int r = 0; r < 8; ++r) {
        int grow = m0 + r + hi * 8;
        int b = grow >> 12, nsp = grow & (NSP - 1);
        out[(size_t)b * CCH * NSP + (size_t)colC * NSP + nsp] = acc[r];
    }
}

extern "C" void kernel_launch(void* const* d_in, const int* in_sizes, int n_in,
                              void* d_out, int out_size, void* d_ws, size_t ws_size,
                              hipStream_t stream) {
    const float* feats    = (const float*)d_in[0];
    const float* codebook = (const float*)d_in[1];
    const float* gumbel   = (const float*)d_in[2];
    float* out = (float*)d_out;

    char* ws = (char*)d_ws;
    float*    sf     = (float*)(ws + WS_SF_OFF);
    float*    sw     = (float*)(ws + WS_SW_OFF);
    _Float16* cbT    = (_Float16*)(ws + WS_CBT_OFF);
    _Float16* soft16 = (_Float16*)(ws + WS_SOFT_OFF);

    sf_kernel<<<NROWS / 256, 256, 0, stream>>>(feats, sf);
    cb_kernel<<<KCW, 256, 0, stream>>>(codebook, sw, cbT, out + KLD_OFF);
    rows_kernel<<<NROWS / 8, 256, 0, stream>>>(gumbel, sf, sw, soft16, out);
    gemm_kernel<<<NROWS / 16, 512, 0, stream>>>(soft16, cbT, out);
}
